// LLModel_88587995447914
// MI455X (gfx1250) — compile-verified, matrix operand order checked by reading, not run on hardware
//
#include <hip/hip_runtime.h>

typedef __attribute__((ext_vector_type(16))) _Float16 v16h;
typedef __attribute__((ext_vector_type(8)))  _Float16 h8;
typedef __attribute__((ext_vector_type(4)))  _Float16 h4;
typedef __attribute__((ext_vector_type(8)))  float    v8f;
typedef __attribute__((ext_vector_type(4)))  float    f4;

union F16x16 { v16h v; h8 h[2]; };
union F32x8  { v8f  v; float f[8]; };

#define BM 128
#define BN 64
#define BK 32
#define PAD 8

// C[z][M,N] = (relu?)( A[z](f32,[M][lda]) * B[z](f16,[N][ldbh]) + bias + resid )
// Assumes M % 128 == 0, K % 32 == 0. N may be ragged (clamped loads, guarded stores).
__global__ __launch_bounds__(256)
void gemm_wmma_f16(const float* __restrict__ A, int lda, long aZ,
                   const _Float16* __restrict__ Bh, int ldbh, long bZ,
                   float* __restrict__ C, int ldc, long cZ,
                   const float* __restrict__ bias,
                   const float* __restrict__ resid,
                   int M, int N, int K, int relu)
{
    __shared__ __align__(16) _Float16 As[2][BM][BK + PAD];
    __shared__ __align__(16) _Float16 Bs[2][BN][BK + PAD];

    const int z = blockIdx.z;
    A  += (size_t)z * aZ;
    Bh += (size_t)z * bZ;
    C  += (size_t)z * cZ;

    const int m0  = blockIdx.y * BM;
    const int n0  = blockIdx.x * BN;
    const int tid = threadIdx.x;
    const int lane = tid & 31, wid = tid >> 5;
    const int wm = wid & 3, wn = wid >> 2;   // 4x2 wave grid, 32x32 per wave
    const int lr = lane & 15, khalf = lane >> 4;

    // B staging: one h8 (16B) per thread per tile
    const int bn = tid >> 2, bkq = tid & 3;
    const int bnc = (n0 + bn < N) ? (n0 + bn) : (N - 1);  // clamped row
    const bool bValid = (n0 + bn) < N;

    f4 aReg[4];
    h8 bReg;
    h8 bzero = {};

    auto loadTile = [&](int k0) {
#pragma unroll
        for (int j = 0; j < 4; ++j) {
            int fid = tid + 256 * j;
            int row = fid >> 3, kq = fid & 7;     // 128 rows x 8 float4-chunks
            aReg[j] = *(const f4*)(A + (size_t)(m0 + row) * lda + k0 + kq * 4);
        }
        bReg = *(const h8*)(Bh + (size_t)bnc * ldbh + k0 + bkq * 8);
        if (!bValid) bReg = bzero;
    };

    auto storeTile = [&](int buf) {
#pragma unroll
        for (int j = 0; j < 4; ++j) {
            int fid = tid + 256 * j;
            int row = fid >> 3, kq = fid & 7;
            h4 hv = { (_Float16)aReg[j].x, (_Float16)aReg[j].y,
                      (_Float16)aReg[j].z, (_Float16)aReg[j].w };
            *(h4*)&As[buf][row][kq * 4] = hv;
        }
        *(h8*)&Bs[buf][bn][bkq * 8] = bReg;
    };

    v8f acc[2][2] = {};
    const int nk = K / BK;

    loadTile(0);
    storeTile(0);
    __syncthreads();

    for (int kt = 0; kt < nk; ++kt) {
        const int cur = kt & 1;
        const bool more = (kt + 1) < nk;
        if (more) loadTile((kt + 1) * BK);   // global loads in flight during WMMA

        F16x16 af[2], bf[2];
#pragma unroll
        for (int mi = 0; mi < 2; ++mi) {
            const _Float16* rp = &As[cur][wm * 32 + mi * 16 + lr][0];
            af[mi].h[0] = *(const h8*)(rp + khalf * 8);
            af[mi].h[1] = *(const h8*)(rp + 16 + khalf * 8);
        }
#pragma unroll
        for (int ni = 0; ni < 2; ++ni) {
            const _Float16* rp = &Bs[cur][wn * 32 + ni * 16 + lr][0];
            bf[ni].h[0] = *(const h8*)(rp + khalf * 8);
            bf[ni].h[1] = *(const h8*)(rp + 16 + khalf * 8);
        }
#pragma unroll
        for (int mi = 0; mi < 2; ++mi)
#pragma unroll
            for (int ni = 0; ni < 2; ++ni)
                acc[mi][ni] = __builtin_amdgcn_wmma_f32_16x16x32_f16(
                    false, af[mi].v, false, bf[ni].v,
                    (short)0, acc[mi][ni], false, false);

        if (more) {
            storeTile(cur ^ 1);
            __syncthreads();
        }
    }

#pragma unroll
    for (int mi = 0; mi < 2; ++mi)
#pragma unroll
        for (int ni = 0; ni < 2; ++ni) {
            F32x8 u; u.v = acc[mi][ni];
#pragma unroll
            for (int r = 0; r < 8; ++r) {
                int grow = m0 + wm * 32 + mi * 16 + r + (lane >= 16 ? 8 : 0);
                int gcol = n0 + wn * 32 + ni * 16 + lr;
                if (gcol < N) {
                    float val = u.f[r];
                    if (bias)  val += bias[gcol];
                    if (resid) val += resid[(size_t)grow * ldc + gcol];
                    if (relu)  val = fmaxf(val, 0.0f);
                    C[(size_t)grow * ldc + gcol] = val;
                }
            }
        }
}

// out[z][c][r] (f16) = in[(z/zDiv)*inZOuter + (z%zDiv)*inZInner + r*inLd + c]
__global__ __launch_bounds__(256)
void transpose_cvt_k(const float* __restrict__ in, int inLd,
                     int zDiv, long inZOuter, long inZInner,
                     _Float16* __restrict__ out, int outLd, long outZ,
                     int R, int Cc)
{
    __shared__ float tile[32][33];
    const int z = blockIdx.z;
    const float* ip = in + (size_t)(z / zDiv) * inZOuter + (size_t)(z % zDiv) * inZInner;
    _Float16* op = out + (size_t)z * outZ;
    const int r0 = blockIdx.y * 32, c0 = blockIdx.x * 32;
    const int tx = threadIdx.x & 31, ty = threadIdx.x >> 5;
    for (int i = ty; i < 32; i += 8) {
        int r = r0 + i, c = c0 + tx;
        tile[i][tx] = (r < R && c < Cc) ? ip[(size_t)r * inLd + c] : 0.0f;
    }
    __syncthreads();
    for (int i = ty; i < 32; i += 8) {
        int oc = c0 + i, orr = r0 + tx;
        if (oc < Cc && orr < R)
            op[(size_t)oc * outLd + orr] = (_Float16)tile[tx][i];
    }
}

// dst[((b*H+h)*T + t)*HS + c] = src[(b*T+t)*(H*HS) + h*HS + c]
__global__ __launch_bounds__(256)
void head_split_f32_k(const float* __restrict__ src, float* __restrict__ dst,
                      int Bb, int T, int H, int HS)
{
    int gid = blockIdx.x * 256 + threadIdx.x;
    int total = Bb * T * H * HS;
    if (gid >= total) return;
    int c = gid % HS;
    int t = (gid / HS) % T;
    int h = (gid / (HS * T)) % H;
    int b = gid / (HS * T * H);
    dst[gid] = src[(size_t)(b * T + t) * (H * HS) + h * HS + c];
}

__global__ __launch_bounds__(256)
void head_split_f16_k(const float* __restrict__ src, _Float16* __restrict__ dst,
                      int Bb, int T, int H, int HS)
{
    int gid = blockIdx.x * 256 + threadIdx.x;
    int total = Bb * T * H * HS;
    if (gid >= total) return;
    int c = gid % HS;
    int t = (gid / HS) % T;
    int h = (gid / (HS * T)) % H;
    int b = gid / (HS * T * H);
    dst[gid] = (_Float16)src[(size_t)(b * T + t) * (H * HS) + h * HS + c];
}

// dst[(b*T+t)*(H*HS) + h*HS + c] = src[((b*H+h)*T + t)*HS + c]
__global__ __launch_bounds__(256)
void head_merge_f32_k(const float* __restrict__ src, float* __restrict__ dst,
                      int Bb, int T, int H, int HS)
{
    int gid = blockIdx.x * 256 + threadIdx.x;
    int total = Bb * T * H * HS;
    if (gid >= total) return;
    int c = gid % HS;
    int h = (gid / HS) % H;
    int t = (gid / (HS * H)) % T;
    int b = gid / (HS * H * T);
    dst[gid] = src[((size_t)(b * H + h) * T + t) * HS + c];
}

__global__ __launch_bounds__(256)
void embed_k(const int* __restrict__ idx, const float* __restrict__ tok,
             const float* __restrict__ pos, float* __restrict__ x,
             int BT, int T, int Dm)
{
    int gid = blockIdx.x * 256 + threadIdx.x;
    int total = BT * Dm;
    if (gid >= total) return;
    int bt = gid / Dm, d = gid - bt * Dm;
    x[gid] = tok[(size_t)idx[bt] * Dm + d] + pos[(size_t)(bt % T) * Dm + d];
}

__global__ __launch_bounds__(256)
void layernorm_k(const float* __restrict__ x, const float* __restrict__ g,
                 const float* __restrict__ b, float* __restrict__ y, int Dm)
{
    __shared__ float s1[256], s2[256];
    const int row = blockIdx.x;
    const float* xr = x + (size_t)row * Dm;
    float sum = 0.f, sq = 0.f;
    for (int d = threadIdx.x; d < Dm; d += 256) { float t = xr[d]; sum += t; sq += t * t; }
    s1[threadIdx.x] = sum; s2[threadIdx.x] = sq;
    __syncthreads();
    for (int s = 128; s > 0; s >>= 1) {
        if (threadIdx.x < s) {
            s1[threadIdx.x] += s1[threadIdx.x + s];
            s2[threadIdx.x] += s2[threadIdx.x + s];
        }
        __syncthreads();
    }
    float mean = s1[0] / Dm;
    float var  = s2[0] / Dm - mean * mean;
    float rstd = rsqrtf(var + 1e-5f);
    float* yr = y + (size_t)row * Dm;
    for (int d = threadIdx.x; d < Dm; d += 256)
        yr[d] = (xr[d] - mean) * rstd * g[d] + b[d];
}

// batched over z = blockIdx.y ; S[z] is [T][T]
__global__ __launch_bounds__(256)
void softmax_causal_k(float* __restrict__ S, int T, float scale)
{
    __shared__ float red[256];
    const int t = blockIdx.x;
    const int z = blockIdx.y;
    float* row = S + (size_t)z * T * T + (size_t)t * T;
    const int nvalid = t + 1;

    float mx = -1e30f;
    for (int s = threadIdx.x; s < nvalid; s += 256) mx = fmaxf(mx, row[s] * scale);
    red[threadIdx.x] = mx; __syncthreads();
    for (int s = 128; s > 0; s >>= 1) {
        if (threadIdx.x < s) red[threadIdx.x] = fmaxf(red[threadIdx.x], red[threadIdx.x + s]);
        __syncthreads();
    }
    mx = red[0]; __syncthreads();

    float sum = 0.f;
    for (int s = threadIdx.x; s < nvalid; s += 256) sum += expf(row[s] * scale - mx);
    red[threadIdx.x] = sum; __syncthreads();
    for (int s = 128; s > 0; s >>= 1) {
        if (threadIdx.x < s) red[threadIdx.x] += red[threadIdx.x + s];
        __syncthreads();
    }
    float inv = 1.0f / red[0];

    for (int s = threadIdx.x; s < T; s += 256)
        row[s] = (s < nvalid) ? expf(row[s] * scale - mx) * inv : 0.0f;
}

__global__ __launch_bounds__(256)
void loss_row_k(const float* __restrict__ logits, const int* __restrict__ tgt,
                float* __restrict__ lossp, int Vv)
{
    __shared__ float red[256];
    const int row = blockIdx.x;
    const float* lrow = logits + (size_t)row * Vv;

    float mx = -1e30f;
    for (int j = threadIdx.x; j < Vv; j += 256) mx = fmaxf(mx, lrow[j]);
    red[threadIdx.x] = mx; __syncthreads();
    for (int s = 128; s > 0; s >>= 1) {
        if (threadIdx.x < s) red[threadIdx.x] = fmaxf(red[threadIdx.x], red[threadIdx.x + s]);
        __syncthreads();
    }
    mx = red[0]; __syncthreads();

    float sum = 0.f;
    for (int j = threadIdx.x; j < Vv; j += 256) sum += expf(lrow[j] - mx);
    red[threadIdx.x] = sum; __syncthreads();
    for (int s = 128; s > 0; s >>= 1) {
        if (threadIdx.x < s) red[threadIdx.x] += red[threadIdx.x + s];
        __syncthreads();
    }
    if (threadIdx.x == 0)
        lossp[row] = -(lrow[tgt[row]] - mx - logf(red[0]));
}

__global__ __launch_bounds__(256)
void loss_reduce_k(const float* __restrict__ lossp, float* __restrict__ out, int n)
{
    __shared__ float red[256];
    float s = 0.f;
    for (int i = threadIdx.x; i < n; i += 256) s += lossp[i];
    red[threadIdx.x] = s; __syncthreads();
    for (int k = 128; k > 0; k >>= 1) {
        if (threadIdx.x < k) red[threadIdx.x] += red[threadIdx.x + k];
        __syncthreads();
    }
    if (threadIdx.x == 0) *out = red[0] / (float)n;
}

extern "C" void kernel_launch(void* const* d_in, const int* in_sizes, int n_in,
                              void* d_out, int out_size, void* d_ws, size_t ws_size,
                              hipStream_t stream)
{
    const int Bb = 2, T = 1024, Dm = 768, H = 12, HS = 64, Lc = 6, FF = 3072, Vv = 50257;
    const int BT = Bb * T;
    const int Z  = Bb * H;     // 24 attention batches
    (void)in_sizes; (void)n_in; (void)out_size; (void)ws_size;

    const int*   idx     = (const int*)d_in[0];
    const int*   targets = (const int*)d_in[1];
    const float* tok_emb = (const float*)d_in[2];
    const float* pos_emb = (const float*)d_in[3];
    const float* ln1_g   = (const float*)d_in[4];
    const float* ln1_b   = (const float*)d_in[5];
    const float* wq      = (const float*)d_in[6];
    const float* wk      = (const float*)d_in[7];
    const float* wv      = (const float*)d_in[8];
    const float* wo      = (const float*)d_in[9];
    const float* bo      = (const float*)d_in[10];
    const float* ln2_g   = (const float*)d_in[11];
    const float* ln2_b   = (const float*)d_in[12];
    const float* w1      = (const float*)d_in[13];
    const float* b1      = (const float*)d_in[14];
    const float* w2      = (const float*)d_in[15];
    const float* b2      = (const float*)d_in[16];
    const float* lnf_g   = (const float*)d_in[17];
    const float* lnf_b   = (const float*)d_in[18];
    const float* w_out   = (const float*)d_in[19];
    const float* b_out   = (const float*)d_in[20];

    float* logits = (float*)d_out;
    float* loss   = logits + (size_t)BT * Vv;

    // workspace carve-up (16B aligned)
    char* p = (char*)d_ws;
    auto allocF = [&](size_t n) { float* r = (float*)p;     p += ((n * 4 + 15) & ~15ULL); return r; };
    auto allocH = [&](size_t n) { _Float16* r = (_Float16*)p; p += ((n * 2 + 15) & ~15ULL); return r; };

    float*    x      = allocF((size_t)BT * Dm);
    float*    xn     = allocF((size_t)BT * Dm);
    float*    qb     = allocF((size_t)BT * Dm);
    float*    kvb    = allocF((size_t)BT * Dm);   // k then v proj output (reused)
    float*    attout = allocF((size_t)BT * Dm);
    float*    orb    = allocF((size_t)Z * T * HS);
    float*    qr     = allocF((size_t)Z * T * HS);
    float*    ffh    = allocF((size_t)BT * FF);
    float*    scb    = allocF((size_t)Z * T * T);
    float*    lossp  = allocF((size_t)BT);
    _Float16* kr     = allocH((size_t)Z * T * HS);
    _Float16* vt     = allocH((size_t)Z * HS * T);
    _Float16* wt     = allocH((size_t)Vv * Dm);   // transposed f16 weight buffer (max = w_out)

    dim3 blk(256);

    auto gemm = [&](const float* A, int lda, long aZ,
                    const _Float16* Bm, int ldbh, long bZ,
                    float* C, int ldc, long cZ,
                    const float* bias, const float* resid,
                    int M, int N, int K, int relu, int zN) {
        dim3 grid((N + BN - 1) / BN, M / BM, zN);
        gemm_wmma_f16<<<grid, blk, 0, stream>>>(A, lda, aZ, Bm, ldbh, bZ,
                                                C, ldc, cZ, bias, resid, M, N, K, relu);
    };
    auto transpose = [&](const float* in, int inLd, int zDiv, long zOut, long zIn,
                         _Float16* out, int outLd, long outZ, int R, int Cc, int zN) {
        dim3 grid((Cc + 31) / 32, (R + 31) / 32, zN);
        transpose_cvt_k<<<grid, blk, 0, stream>>>(in, inLd, zDiv, zOut, zIn,
                                                  out, outLd, outZ, R, Cc);
    };

    const int elemsBTD = BT * Dm;
    const int gBTD = (elemsBTD + 255) / 256;

    embed_k<<<gBTD, blk, 0, stream>>>(idx, tok_emb, pos_emb, x, BT, T, Dm);

    const float scale = 0.125f;  // HS^-0.5

    for (int l = 0; l < Lc; ++l) {
        layernorm_k<<<BT, blk, 0, stream>>>(x, ln1_g + l * Dm, ln1_b + l * Dm, xn, Dm);

        // Q = xn @ wq[l]  (wq[l]: [H][D][HS] -> wt [H*HS][D])
        transpose(wq + (size_t)l * H * Dm * HS, HS, 1, (long)Dm * HS, 0, wt, Dm, (long)HS * Dm, Dm, HS, H);
        gemm(xn, Dm, 0, wt, Dm, 0, qb, Dm, 0, nullptr, nullptr, BT, Dm, Dm, 0, 1);
        head_split_f32_k<<<gBTD, blk, 0, stream>>>(qb, qr, Bb, T, H, HS);

        // K
        transpose(wk + (size_t)l * H * Dm * HS, HS, 1, (long)Dm * HS, 0, wt, Dm, (long)HS * Dm, Dm, HS, H);
        gemm(xn, Dm, 0, wt, Dm, 0, kvb, Dm, 0, nullptr, nullptr, BT, Dm, Dm, 0, 1);
        head_split_f16_k<<<gBTD, blk, 0, stream>>>(kvb, kr, Bb, T, H, HS);

        // V  (then transpose per (b,h) into vt [z][HS][T])
        transpose(wv + (size_t)l * H * Dm * HS, HS, 1, (long)Dm * HS, 0, wt, Dm, (long)HS * Dm, Dm, HS, H);
        gemm(xn, Dm, 0, wt, Dm, 0, kvb, Dm, 0, nullptr, nullptr, BT, Dm, Dm, 0, 1);
        transpose(kvb, Dm, H, (long)T * Dm, (long)HS, vt, T, (long)HS * T, T, HS, Z);

        // S = Q K^T  (z = 24), causal softmax, O = P V
        gemm(qr, HS, (long)T * HS, kr, HS, (long)T * HS, scb, T, (long)T * T,
             nullptr, nullptr, T, T, HS, 0, Z);
        {
            dim3 g(T, Z);
            softmax_causal_k<<<g, blk, 0, stream>>>(scb, T, scale);
        }
        gemm(scb, T, (long)T * T, vt, T, (long)HS * T, orb, HS, (long)T * HS,
             nullptr, nullptr, T, HS, T, 0, Z);
        head_merge_f32_k<<<gBTD, blk, 0, stream>>>(orb, attout, Bb, T, H, HS);

        // x = x + attout @ wo + bo
        transpose(wo + (size_t)l * Dm * Dm, Dm, 1, 0, 0, wt, Dm, 0, Dm, Dm, 1);
        gemm(attout, Dm, 0, wt, Dm, 0, x, Dm, 0, bo + l * Dm, x, BT, Dm, Dm, 0, 1);

        layernorm_k<<<BT, blk, 0, stream>>>(x, ln2_g + l * Dm, ln2_b + l * Dm, xn, Dm);

        // ffh = relu(xn @ w1 + b1) ; x = x + ffh @ w2 + b2
        transpose(w1 + (size_t)l * Dm * FF, FF, 1, 0, 0, wt, Dm, 0, Dm, FF, 1);
        gemm(xn, Dm, 0, wt, Dm, 0, ffh, FF, 0, b1 + l * FF, nullptr, BT, FF, Dm, 1, 1);
        transpose(w2 + (size_t)l * FF * Dm, Dm, 1, 0, 0, wt, FF, 0, FF, Dm, 1);
        gemm(ffh, FF, 0, wt, FF, 0, x, Dm, 0, b2 + l * Dm, x, BT, Dm, FF, 0, 1);
    }

    layernorm_k<<<BT, blk, 0, stream>>>(x, lnf_g, lnf_b, xn, Dm);

    // logits = xn @ w_out + b_out   (w_out: [D][V] -> wt [V][D], ragged V)
    transpose(w_out, Vv, 1, 0, 0, wt, Dm, 0, Dm, Vv, 1);
    gemm(xn, Dm, 0, wt, Dm, 0, logits, Vv, 0, b_out, nullptr, BT, Vv, Dm, 0, 1);

    loss_row_k<<<BT, blk, 0, stream>>>(logits, targets, lossp, Vv);
    loss_reduce_k<<<1, blk, 0, stream>>>(lossp, loss, BT);
}
